// SelfAttentionBlock1d_3444563772214
// MI455X (gfx1250) — matrix operand order repaired and checked by reading, not compiled, not run on hardware
//
#include <hip/hip_runtime.h>

// B=8, C=512, L=2048, E=512 self-attention block.
// f16 WMMA (16x16x32) + f32 accumulate, 64x64 register-blocked per wave.
// Bandwidth kernels (softmax row staging, transpose fill) use the CDNA5
// async global->LDS path (ASYNCcnt) instead of VGPR round-trips.

typedef __attribute__((ext_vector_type(16))) _Float16 v16h;
typedef __attribute__((ext_vector_type(8)))  _Float16 v8h;
typedef __attribute__((ext_vector_type(8)))  float    v8f;

union FragU { v16h v; v8h h[2]; };

// ---- CDNA5 async global->LDS helpers (ISA 10.7 / 15.18.3) ------------------
// Generic pointers into __shared__ have the wave-relative LDS byte offset in
// their low 32 bits (flat-LDS aperture mapping), which is exactly the VDST
// operand of the async-load instructions.
__device__ __forceinline__ void async_ld_lds_b128(void* lds, const void* g) {
  unsigned l = (unsigned)(unsigned long long)lds;
  asm volatile("global_load_async_to_lds_b128 %0, %1, off"
               :: "v"(l), "v"(g) : "memory");
}
__device__ __forceinline__ void async_ld_lds_b32(void* lds, const void* g) {
  unsigned l = (unsigned)(unsigned long long)lds;
  asm volatile("global_load_async_to_lds_b32 %0, %1, off"
               :: "v"(l), "v"(g) : "memory");
}
__device__ __forceinline__ void wait_async0() {
#if __has_builtin(__builtin_amdgcn_s_wait_asynccnt)
  __builtin_amdgcn_s_wait_asynccnt(0);
#else
  asm volatile("s_wait_asynccnt 0x0" ::: "memory");
#endif
}

// ---- Register-blocked batched WMMA GEMM ------------------------------------
// C[m,n] = sum_k A[m,k] * Bt[n,k] (+bias)(+resid); each wave owns a 64x64 tile.
// A row-major [M,K]; Bt row-major [N,K] (i.e. B transposed); M,N % 64 == 0,
// K % 32 == 0. BIAS_MODE: 0 = none, 1 = bias[m], 2 = bias[n].
template <int BIAS_MODE, bool OUT_F16, bool RESID>
__global__ __launch_bounds__(256) void wmma_gemm64(
    const _Float16* __restrict__ A, int lda, long strideA,
    const _Float16* __restrict__ Bt, int ldb, long strideB,
    void* __restrict__ Cout, int ldc, long strideC,
    const float* __restrict__ bias,
    const float* __restrict__ resid, long strideR,
    int M, int N, int K) {
  const int b    = blockIdx.y;
  const int lane = threadIdx.x & 31;
  const int wave = threadIdx.x >> 5;
  const int tilesN = N >> 6;
  const int tilesM = M >> 6;
  const int tile = blockIdx.x * 8 + wave;
  if (tile >= tilesM * tilesN) return;   // uniform per wave: EXEC stays full
  const int tm = tile / tilesN;
  const int tn = tile - tm * tilesN;
  const int m0 = tm << 6;
  const int n0 = tn << 6;

  const _Float16* Ab = A  + (size_t)b * strideA;
  const _Float16* Bb = Bt + (size_t)b * strideB;

  const int rlane = lane & 15;                 // row/col within subtile
  const int aoff  = (lane & 16) ? 8  : 0;      // A fragment K offset
  const int boff  = (lane & 16) ? 8  : 0;      // B fragment K offset
  const int mOff  = (lane & 16) ? 8  : 0;      // C/D row offset

  // Accumulators (+ bias pre-load).
  v8f acc[4][4];
#pragma unroll
  for (int i = 0; i < 4; ++i)
#pragma unroll
    for (int j = 0; j < 4; ++j) {
      v8f a = {};
      if (BIAS_MODE == 2) {
        const float bv = bias[n0 + (j << 4) + rlane];
#pragma unroll
        for (int r = 0; r < 8; ++r) a[r] = bv;
      } else if (BIAS_MODE == 1) {
#pragma unroll
        for (int r = 0; r < 8; ++r) a[r] = bias[m0 + (i << 4) + mOff + r];
      }
      acc[i][j] = a;
    }

  // Per-lane fragment base pointers (fixed row per lane).
  const _Float16* aptr[4];
  const _Float16* bptr[4];
#pragma unroll
  for (int i = 0; i < 4; ++i)
    aptr[i] = Ab + (size_t)(m0 + (i << 4) + rlane) * lda + aoff;
#pragma unroll
  for (int j = 0; j < 4; ++j)
    bptr[j] = Bb + (size_t)(n0 + (j << 4) + rlane) * ldb + boff;

  for (int k0 = 0; k0 < K; k0 += 32) {
    if (k0 + 32 < K) {   // prefetch next K-slice (global_prefetch_b8)
      __builtin_prefetch(aptr[0] + k0 + 32, 0, 1);
      __builtin_prefetch(bptr[0] + k0 + 32, 0, 1);
    }
    v16h af[4], bf[4];
#pragma unroll
    for (int i = 0; i < 4; ++i) {
      FragU f;
      f.h[0] = *(const v8h*)(aptr[i] + k0);
      f.h[1] = *(const v8h*)(aptr[i] + k0 + 16);
      af[i] = f.v;
    }
#pragma unroll
    for (int j = 0; j < 4; ++j) {
      FragU f;
      f.h[0] = *(const v8h*)(bptr[j] + k0);
      f.h[1] = *(const v8h*)(bptr[j] + k0 + 16);
      bf[j] = f.v;
    }
#pragma unroll
    for (int i = 0; i < 4; ++i)
#pragma unroll
      for (int j = 0; j < 4; ++j)
        acc[i][j] = __builtin_amdgcn_wmma_f32_16x16x32_f16(
            false, af[i], false, bf[j], (short)0, acc[i][j], false, false);
  }

  // Store 64x64 tile.
#pragma unroll
  for (int i = 0; i < 4; ++i) {
    const int mBase = m0 + (i << 4) + mOff;
#pragma unroll
    for (int j = 0; j < 4; ++j) {
      const int n = n0 + (j << 4) + rlane;
      if (OUT_F16) {
        _Float16* C = (_Float16*)Cout + (size_t)b * strideC;
#pragma unroll
        for (int r = 0; r < 8; ++r)
          C[(size_t)(mBase + r) * ldc + n] = (_Float16)acc[i][j][r];
      } else {
        float* C = (float*)Cout + (size_t)b * strideC;
        const float* Rb = RESID ? (resid + (size_t)b * strideR) : nullptr;
#pragma unroll
        for (int r = 0; r < 8; ++r) {
          float v = acc[i][j][r];
          if (RESID) v += Rb[(size_t)(mBase + r) * ldc + n];
          C[(size_t)(mBase + r) * ldc + n] = v;
        }
      }
    }
  }
}

// ---- x [B,C,L] f32 -> Xt [B,L,C] f16, async-fill LDS-tiled transpose -------
__global__ __launch_bounds__(256) void transpose_cast(
    const float* __restrict__ x, _Float16* __restrict__ Xt, int C, int L) {
  __shared__ float tile[32][33];
  const int b  = blockIdx.z;
  const int c0 = blockIdx.y << 5;
  const int l0 = blockIdx.x << 5;
  const int tx = threadIdx.x;
  const float* xb = x + (size_t)b * C * L;
  // Fill phase: async global->LDS, no VGPR round-trip.
  for (int i = threadIdx.y; i < 32; i += 8)
    async_ld_lds_b32(&tile[i][tx], &xb[(size_t)(c0 + i) * L + l0 + tx]);
  wait_async0();
  __syncthreads();
  _Float16* xt = Xt + (size_t)b * L * C;
  for (int i = threadIdx.y; i < 32; i += 8)
    xt[(size_t)(l0 + i) * C + c0 + tx] = (_Float16)tile[tx][i];
}

// ---- f32 -> f16 cast for weights ------------------------------------------
__global__ __launch_bounds__(256) void cast_f32_f16(
    const float* __restrict__ in, _Float16* __restrict__ out, int n) {
  int i = blockIdx.x * 256 + threadIdx.x;
  if (i < n) out[i] = (_Float16)in[i];
}

// ---- row softmax: S f32 [rows, 2048] -> P f16, block 256 -------------------
// Row is staged to LDS with async b128 transfers, reductions run from LDS.
__global__ __launch_bounds__(256) void softmax_row(
    const float* __restrict__ S, _Float16* __restrict__ P, int Lc) {
  __shared__ float srow[2048];
  __shared__ float red[8];
  const size_t row = blockIdx.x;
  const float* s = S + row * Lc;
  _Float16*    p = P + row * Lc;
  const int t = threadIdx.x;

  // Stage 8KB row: each thread moves 2x16B via the async engine.
  async_ld_lds_b128(&srow[4 * t],        &s[4 * t]);
  async_ld_lds_b128(&srow[1024 + 4 * t], &s[1024 + 4 * t]);
  wait_async0();
  __syncthreads();

  float ev[8];
  float m = -3.402823466e38f;
#pragma unroll
  for (int i = 0; i < 8; ++i) {
    ev[i] = srow[t + (i << 8)];
    m = fmaxf(m, ev[i]);
  }
  for (int off = 16; off > 0; off >>= 1) m = fmaxf(m, __shfl_xor(m, off, 32));
  if ((t & 31) == 0) red[t >> 5] = m;
  __syncthreads();
  if (t < 32) {
    float v2 = (t < 8) ? red[t] : -3.402823466e38f;
    for (int off = 4; off > 0; off >>= 1) v2 = fmaxf(v2, __shfl_xor(v2, off, 32));
    if (t == 0) red[0] = v2;
  }
  __syncthreads();
  m = red[0];

  float sum = 0.f;
#pragma unroll
  for (int i = 0; i < 8; ++i) {
    ev[i] = __expf(ev[i] - m);
    sum += ev[i];
  }
  for (int off = 16; off > 0; off >>= 1) sum += __shfl_xor(sum, off, 32);
  __syncthreads();
  if ((t & 31) == 0) red[t >> 5] = sum;
  __syncthreads();
  if (t < 32) {
    float v2 = (t < 8) ? red[t] : 0.f;
    for (int off = 4; off > 0; off >>= 1) v2 += __shfl_xor(v2, off, 32);
    if (t == 0) red[0] = v2;
  }
  __syncthreads();
  const float inv = 1.0f / red[0];
#pragma unroll
  for (int i = 0; i < 8; ++i) p[t + (i << 8)] = (_Float16)(ev[i] * inv);
}

extern "C" void kernel_launch(void* const* d_in, const int* in_sizes, int n_in,
                              void* d_out, int out_size, void* d_ws, size_t ws_size,
                              hipStream_t stream) {
  (void)in_sizes; (void)n_in; (void)out_size; (void)ws_size;
  const int Bb = 8, Cc = 512, Ll = 2048, Ee = 512;

  const float* x  = (const float*)d_in[0];
  const float* wq = (const float*)d_in[1];
  const float* bq = (const float*)d_in[2];
  const float* wk = (const float*)d_in[3];
  const float* bk = (const float*)d_in[4];
  const float* wv = (const float*)d_in[5];
  const float* bv = (const float*)d_in[6];
  const float* wo = (const float*)d_in[7];
  const float* bo = (const float*)d_in[8];
  float* y = (float*)d_out;

  char* ws = (char*)d_ws;
  size_t off = 0;
  _Float16* Xt  = (_Float16*)(ws + off); off += (size_t)Bb * Ll * Cc * 2;  // x^T f16
  _Float16* Qt  = (_Float16*)(ws + off); off += (size_t)Bb * Ll * Ee * 2;  // q^T [B,L,E]
  _Float16* Kt  = (_Float16*)(ws + off); off += (size_t)Bb * Ll * Ee * 2;  // k^T [B,L,E]
  _Float16* Vm  = (_Float16*)(ws + off); off += (size_t)Bb * Ee * Ll * 2;  // v   [B,E,L]
  _Float16* Ot  = (_Float16*)(ws + off); off += (size_t)Bb * Ll * Ee * 2;  // out [B,L,E]
  _Float16* Wq16 = (_Float16*)(ws + off); off += (size_t)Ee * Cc * 2;
  _Float16* Wk16 = (_Float16*)(ws + off); off += (size_t)Ee * Cc * 2;
  _Float16* Wv16 = (_Float16*)(ws + off); off += (size_t)Ee * Cc * 2;
  _Float16* Wo16 = (_Float16*)(ws + off); off += (size_t)Cc * Ee * 2;
  float*    Sc  = (float*)(ws + off);    off += (size_t)Bb * Ll * Ll * 4;  // scores f32
  _Float16* At  = (_Float16*)(ws + off); off += (size_t)Bb * Ll * Ll * 2;  // attn f16

  // 1) weights f32 -> f16
  {
    const int n = Ee * Cc;
    const int g = (n + 255) / 256;
    cast_f32_f16<<<g, 256, 0, stream>>>(wq, Wq16, n);
    cast_f32_f16<<<g, 256, 0, stream>>>(wk, Wk16, n);
    cast_f32_f16<<<g, 256, 0, stream>>>(wv, Wv16, n);
    cast_f32_f16<<<g, 256, 0, stream>>>(wo, Wo16, n);
  }

  // 2) x -> Xt (transpose + cast)
  transpose_cast<<<dim3(Ll / 32, Cc / 32, Bb), dim3(32, 8), 0, stream>>>(x, Xt, Cc, Ll);

  const long sXt = (long)Ll * Cc;
  const long sLE = (long)Ll * Ee;
  const long sEL = (long)Ee * Ll;
  const long sLL = (long)Ll * Ll;
  const long sCL = (long)Cc * Ll;

  // 3) Qt[l,e] = Xt[l,:]·wq[e,:] + bq[e]   (M=L, N=E, K=C; bias over N)
  {
    const int tiles = (Ll / 64) * (Ee / 64);
    wmma_gemm64<2, true, false><<<dim3((tiles + 7) / 8, Bb), 256, 0, stream>>>(
        Xt, Cc, sXt, Wq16, Cc, 0, Qt, Ee, sLE, bq, nullptr, 0, Ll, Ee, Cc);
  // 4) Kt likewise
    wmma_gemm64<2, true, false><<<dim3((tiles + 7) / 8, Bb), 256, 0, stream>>>(
        Xt, Cc, sXt, Wk16, Cc, 0, Kt, Ee, sLE, bk, nullptr, 0, Ll, Ee, Cc);
  }

  // 5) V[e,l] = wv[e,:]·Xt[l,:] + bv[e]    (M=E, N=L, K=C; bias over M)
  {
    const int tiles = (Ee / 64) * (Ll / 64);
    wmma_gemm64<1, true, false><<<dim3((tiles + 7) / 8, Bb), 256, 0, stream>>>(
        Wv16, Cc, 0, Xt, Cc, sXt, Vm, Ll, sEL, bv, nullptr, 0, Ee, Ll, Cc);
  }

  // 6) scores[i,j] = Qt[i,:]·Kt[j,:]       (M=L, N=L, K=E; f32 out)
  {
    const int tiles = (Ll / 64) * (Ll / 64);
    wmma_gemm64<0, false, false><<<dim3((tiles + 7) / 8, Bb), 256, 0, stream>>>(
        Qt, Ee, sLE, Kt, Ee, sLE, Sc, Ll, sLL, nullptr, nullptr, 0, Ll, Ll, Ee);
  }

  // 7) row softmax -> f16 probabilities
  softmax_row<<<Bb * Ll, 256, 0, stream>>>(Sc, At, Ll);

  // 8) Out[i,e] = attn[i,:]·V[e,:]         (M=L, N=E, K=L)
  {
    const int tiles = (Ll / 64) * (Ee / 64);
    wmma_gemm64<0, true, false><<<dim3((tiles + 7) / 8, Bb), 256, 0, stream>>>(
        At, Ll, sLL, Vm, Ll, sEL, Ot, Ee, sLE, nullptr, nullptr, 0, Ll, Ee, Ll);
  }

  // 9) y[c,l] = x[c,l] + wo[c,:]·Out[l,:] + bo[c]  (M=C, N=L, K=E; f32 out + resid)
  {
    const int tiles = (Cc / 64) * (Ll / 64);
    wmma_gemm64<1, false, true><<<dim3((tiles + 7) / 8, Bb), 256, 0, stream>>>(
        Wo16, Ee, 0, Ot, Ee, sLE, y, Ll, sCL, bo, x, sCL, Cc, Ll, Ee);
  }
}
